// SpikingLayer_12008728560082
// MI455X (gfx1250) — compile-verified, matrix operand order checked by reading, not run on hardware
//
#include <hip/hip_runtime.h>
#include <hip/hip_bf16.h>
#include <math.h>

// ---------------------------------------------------------------------------
// SpikingLayer for MI455X (gfx1250, wave32, WMMA, TDM).
//   K_init     : zero h ping-pong + grid barrier counter
//   K_cvt      : f32 -> bf16 weight conversion (W_ih)
//   K_prep_whh : W_hh -> per-WG permuted, padded bf16 strips (TDM-friendly)
//   K_wc/K_bc  : fold linear into mixer: Wc = mix_w[:, :H] @ lin_w, bc fold
//   K_xg       : xg = x @ W_ih^T + b_ih + b_hh   (bf16 WMMA, bf16 out)
//   K_scan     : persistent LSTM scan; W_hh strip TDM-loaded into 266KB LDS
//                once, reused for all 2048 steps; c in registers; one grid
//                barrier per step; h ping-pong in bf16
//   K_mix      : out = Dyt(spk @ Wc^T + x @ W2^T + bc)   (bf16 WMMA)
// ---------------------------------------------------------------------------

#define DEV_ __device__ __forceinline__

typedef __attribute__((ext_vector_type(16))) __bf16 v16bf;
typedef __attribute__((ext_vector_type(8)))  float  v8f;

typedef __attribute__((ext_vector_type(4))) unsigned int tdm_g0_t;
typedef __attribute__((ext_vector_type(8))) int          tdm_g1_t;
typedef __attribute__((ext_vector_type(4))) int          tdm_g2_t;
#if defined(__clang_major__) && (__clang_major__ >= 23)
typedef __attribute__((ext_vector_type(8))) int          tdm_g4_t;
#endif

constexpr int Bsz  = 32;
constexpr int Ssz  = 2048;
constexpr int Dsz  = 1024;
constexpr int Hsz  = 1024;
constexpr int Gsz  = 4 * Hsz;        // 4096 gate columns
constexpr int Mtot = Bsz * Ssz;      // 65536 rows

constexpr int NWG     = 32;          // persistent scan workgroups
constexpr int HID_WG  = Hsz / NWG;   // 32 hidden cols per WG
constexpr int ROWS_WG = 4 * HID_WG;  // 128 gate rows per WG
constexpr int WLD     = 1024 + 16;   // padded K stride (bf16 elems) for strips

// ---- workspace layout (bytes) ----
constexpr size_t SZ_XG   = (size_t)Mtot * Gsz * 2;            // 536,870,912
constexpr size_t SZ_SPK  = (size_t)Mtot * Hsz * 2;            // 134,217,728
constexpr size_t SZ_W4   = (size_t)Gsz * Dsz * 2;             //   8,388,608
constexpr size_t SZ_WHHP = (size_t)NWG * ROWS_WG * WLD * 2;   //   8,519,680
constexpr size_t SZ_W1   = (size_t)Hsz * Hsz * 2;             //   2,097,152
constexpr size_t OFF_XG  = 0;
constexpr size_t OFF_SPK = OFF_XG  + SZ_XG;
constexpr size_t OFF_WIH = OFF_SPK + SZ_SPK;
constexpr size_t OFF_WHH = OFF_WIH + SZ_W4;
constexpr size_t OFF_WC  = OFF_WHH + SZ_WHHP;
constexpr size_t OFF_W2  = OFF_WC  + SZ_W1;
constexpr size_t OFF_BC  = OFF_W2  + SZ_W1;          // 1024 f32
constexpr size_t OFF_H   = OFF_BC  + 4096;           // 2 x 32 x 1024 bf16 ping-pong
constexpr size_t OFF_BAR = OFF_H   + 131072;
constexpr size_t WS_NEED = OFF_BAR + 256;

// dynamic LDS for K_scan: weight strip + gate buffer (283,136 B < 320KB/WG)
constexpr unsigned SCAN_SMEM_WSH = (unsigned)(ROWS_WG * WLD * 2);     // 266,240
constexpr unsigned SCAN_SMEM     = SCAN_SMEM_WSH + 32 * 132 * 4;      // +16,896

// ---- WMMA helpers -----------------------------------------------------------
union FragBF { v16bf v; uint4 q[2]; };

// 16x32 bf16 fragment (A rows / B cols share K-striping): lane l holds
// row/col (l&15); K packs as k = 8*(l>>4) + {0..7} (q[0]) and +16 (q[1]).
DEV_ v16bf ld_frag_k(const __bf16* __restrict__ p0, int ld_elems) {
  const int lane = (int)(threadIdx.x & 31);
  const __bf16* p = p0 + (size_t)(lane & 15) * ld_elems + ((lane >> 4) << 3);
  FragBF f;
  f.q[0] = *(const uint4*)(p);
  f.q[1] = *(const uint4*)(p + 16);
  return f.v;
}

DEV_ v8f wmma_bf16(v16bf a, v16bf b, v8f c) {
  return __builtin_amdgcn_wmma_f32_16x16x32_bf16(
      /*neg_a=*/false, a, /*neg_b=*/false, b,
      /*c_mod=*/(short)0, c, /*reuse_a=*/false, /*reuse_b=*/false);
}

DEV_ float sigmoidf_(float x) { return 1.0f / (1.0f + __expf(-x)); }

// Issue one 2D TDM tile load: 128 rows x 1040 bf16, contiguous tensor
// (stride == width), into LDS offset 0 of this workgroup's allocation.
// D# per CDNA5 ISA ch.8: group0 {count=1, lds_addr, global_addr, type=2},
// group1 {data_size=2B, tensor_dim 1040x128, tile_dim 1040x128, stride 1040}.
DEV_ void tdm_load_strip(const __bf16* gsrc) {
  const unsigned long long ga = (unsigned long long)gsrc;
  tdm_g0_t g0;
  g0[0] = 1u;                                   // count = 1 (valid descriptor)
  g0[1] = 0u;                                   // lds_addr = 0 (dyn-LDS base)
  g0[2] = (unsigned)(ga & 0xffffffffu);         // global_addr[31:0]
  g0[3] = (unsigned)(ga >> 32) | 0x80000000u;   // global_addr[56:32] | type=2
  tdm_g1_t g1;
  g1[0] = 0x00010000;                           // mask=0, data_size=1 (2B)
  g1[1] = (int)(WLD << 16);                     // tensor_dim0 = 1040 (lo16)
  g1[2] = (int)(ROWS_WG << 16);                 // tensor_dim1 = 128 (lo16)
  g1[3] = (int)(WLD << 16);                     // tile_dim0 = 1040
  g1[4] = ROWS_WG;                              // tile_dim1 = 128, tile_dim2=0
  g1[5] = WLD;                                  // tensor_dim0_stride = 1040
  g1[6] = 0;
  g1[7] = 0;
  tdm_g2_t gz = {0, 0, 0, 0};                   // groups 2/3 unused (2D tile)
#if defined(__clang_major__) && (__clang_major__ >= 23)
  tdm_g4_t gz8 = {0, 0, 0, 0, 0, 0, 0, 0};
  __builtin_amdgcn_tensor_load_to_lds(g0, g1, gz, gz, gz8, 0);
#else
  __builtin_amdgcn_tensor_load_to_lds(g0, g1, gz, gz, 0);
#endif
}

// ---- init -------------------------------------------------------------------
__global__ void K_init(__bf16* __restrict__ hbf, unsigned* __restrict__ bar) {
  const int i = blockIdx.x * blockDim.x + threadIdx.x;
  if (i < 2 * Bsz * Hsz) hbf[i] = (__bf16)0.0f;
  if (i == 0) *bar = 0u;
}

// ---- f32 -> bf16 conversions -----------------------------------------------
__global__ void K_cvt(const float* __restrict__ in, __bf16* __restrict__ out, int n) {
  int i = blockIdx.x * blockDim.x + threadIdx.x;
  const int stride = gridDim.x * blockDim.x;
  for (; i < n; i += stride) out[i] = (__bf16)in[i];
}

// W_hh -> per-WG contiguous strips: Whh_p[wg][r = q*32+jl][k], stride WLD,
// so each WG's 128x1040 strip is one contiguous, TDM-loadable 2D tensor.
__global__ void K_prep_whh(const float* __restrict__ Whh, __bf16* __restrict__ Whh_p) {
  const int i = blockIdx.x * blockDim.x + threadIdx.x;   // over 32*128*1024
  if (i >= NWG * ROWS_WG * 1024) return;
  const int wg = i >> 17;
  const int r  = (i >> 10) & (ROWS_WG - 1);
  const int k  = i & 1023;
  const int q  = r >> 5, jl = r & 31;
  Whh_p[(size_t)wg * ROWS_WG * WLD + (size_t)r * WLD + k] =
      (__bf16)Whh[(size_t)(q * Hsz + wg * HID_WG + jl) * Hsz + k];
}

__global__ void K_cvt_mixw2(const float* __restrict__ mixw, __bf16* __restrict__ W2) {
  const int i = blockIdx.x * blockDim.x + threadIdx.x;   // over H*D
  if (i >= Hsz * Dsz) return;
  const int n = i >> 10, d = i & 1023;
  W2[i] = (__bf16)mixw[(size_t)n * (Hsz + Dsz) + Hsz + d];
}

// ---- Wc = mix_w[:, :H] @ lin_w  (f32 accum, bf16 out); tiny one-time GEMM ---
__global__ void __launch_bounds__(256)
K_wc(const float* __restrict__ mixw, const float* __restrict__ linw,
     __bf16* __restrict__ Wc) {
  __shared__ float Ms[16][17];
  __shared__ float Ls[16][17];
  const int tx = threadIdx.x & 15, ty = threadIdx.x >> 4;
  const int n = blockIdx.y * 16 + ty;
  const int h = blockIdx.x * 16 + tx;
  float acc = 0.0f;
  for (int kb = 0; kb < Hsz / 16; ++kb) {
    Ms[ty][tx] = mixw[(size_t)n * (Hsz + Dsz) + kb * 16 + tx];
    Ls[ty][tx] = linw[(size_t)(kb * 16 + ty) * Hsz + h];
    __syncthreads();
#pragma unroll
    for (int k = 0; k < 16; ++k) acc += Ms[ty][k] * Ls[k][tx];
    __syncthreads();
  }
  Wc[(size_t)n * Hsz + h] = (__bf16)acc;
}

__global__ void K_bc(const float* __restrict__ mixw, const float* __restrict__ linb,
                     const float* __restrict__ mixb, float* __restrict__ bc) {
  const int n = blockIdx.x * blockDim.x + threadIdx.x;
  if (n >= Hsz) return;
  float s = mixb[n];
  for (int p = 0; p < Hsz; ++p) s += mixw[(size_t)n * (Hsz + Dsz) + p] * linb[p];
  bc[n] = s;
}

// ---- xg = x @ W_ih^T + b_ih + b_hh  ->  bf16 [Mtot, 4H] ---------------------
__global__ void __launch_bounds__(256)
K_xg(const float* __restrict__ x, const __bf16* __restrict__ Wih,
     const float* __restrict__ b_ih, const float* __restrict__ b_hh,
     __bf16* __restrict__ xg) {
  __shared__ __bf16 As[128 * 40];
  __shared__ __bf16 Bs[128 * 40];
  __shared__ float  bsum[128];

  const int tid = threadIdx.x;
  const int m0 = blockIdx.x * 128;
  const int n0 = blockIdx.y * 128;
  if (tid < 128) bsum[tid] = b_ih[n0 + tid] + b_hh[n0 + tid];

  const int wave = tid >> 5;
  const int wm = wave & 3;
  const int wn = wave >> 2;
  const int lane = tid & 31;

  const v8f vzero = {0.f, 0.f, 0.f, 0.f, 0.f, 0.f, 0.f, 0.f};
  v8f acc[2][4];
#pragma unroll
  for (int i = 0; i < 2; ++i)
#pragma unroll
    for (int j = 0; j < 4; ++j) acc[i][j] = vzero;

  for (int kt = 0; kt < Dsz / 32; ++kt) {
    const int k0 = kt * 32;
#pragma unroll
    for (int i = 0; i < 4; ++i) {            // A: 128x32 f32 -> bf16
      const int s = tid + i * 256;
      const int row = s >> 3, kk = (s & 7) * 4;
      const float4 v = *(const float4*)(x + (size_t)(m0 + row) * Dsz + k0 + kk);
      union { __bf16 h[4]; uint2 u; } pk;
      pk.h[0] = (__bf16)v.x; pk.h[1] = (__bf16)v.y;
      pk.h[2] = (__bf16)v.z; pk.h[3] = (__bf16)v.w;
      *(uint2*)(As + row * 40 + kk) = pk.u;
    }
#pragma unroll
    for (int i = 0; i < 2; ++i) {            // B: 128x32 bf16 (b128 both sides)
      const int s = tid + i * 256;
      const int row = s >> 2, kk = (s & 3) * 8;
      *(uint4*)(Bs + row * 40 + kk) =
          *(const uint4*)(Wih + (size_t)(n0 + row) * Dsz + k0 + kk);
    }
    __syncthreads();
    v16bf a[2], b[4];
#pragma unroll
    for (int i = 0; i < 2; ++i) a[i] = ld_frag_k(As + (wm * 32 + i * 16) * 40, 40);
#pragma unroll
    for (int j = 0; j < 4; ++j) b[j] = ld_frag_k(Bs + (wn * 64 + j * 16) * 40, 40);
#pragma unroll
    for (int i = 0; i < 2; ++i)
#pragma unroll
      for (int j = 0; j < 4; ++j) acc[i][j] = wmma_bf16(a[i], b[j], acc[i][j]);
    __syncthreads();
  }
#pragma unroll
  for (int i = 0; i < 2; ++i)
#pragma unroll
    for (int j = 0; j < 4; ++j) {
      const int mb = m0 + wm * 32 + i * 16 + (lane >> 4) * 8;
      const int nl = wn * 64 + j * 16 + (lane & 15);
      const float bias = bsum[nl];
#pragma unroll
      for (int v = 0; v < 8; ++v)
        xg[(size_t)(mb + v) * Gsz + (n0 + nl)] = (__bf16)(acc[i][j][v] + bias);
    }
}

// ---- persistent LSTM scan ---------------------------------------------------
// 32 WGs x 256 thr; WG owns 32 hidden cols (128 gate rows). The WG's W_hh
// strip (128x1040 bf16, 266KB) is TDM-loaded into LDS ONCE, then reused for
// all 2048 steps: inner loop = 1 global h-frag + 2 LDS b-frags + 2 WMMA.
__global__ void __launch_bounds__(256)
K_scan(const __bf16* __restrict__ xg, const __bf16* __restrict__ Whh_p,
       __bf16* __restrict__ hbf, __bf16* __restrict__ spk,
       const float* __restrict__ thr_p, unsigned* __restrict__ bar, int nwg) {
  extern __shared__ char smem[];
  __bf16* Wsh = (__bf16*)smem;                       // [128][WLD]
  float*  Gs  = (float*)(smem + SCAN_SMEM_WSH);      // [32][132]

  const int tid  = threadIdx.x;
  const int wave = tid >> 5;
  const int lane = tid & 31;
  const int wm = wave & 1;           // M half (rows 16*wm..)
  const int q  = wave >> 1;          // gate index 0..3
  const int hid0 = blockIdx.x * HID_WG;
  const float thr = thr_p[0];

  // one-shot Tensor Data Mover load of this WG's weight strip into LDS
  if (wave == 0) {
    tdm_load_strip(Whh_p + (size_t)blockIdx.x * ROWS_WG * WLD);
    __builtin_amdgcn_s_wait_tensorcnt((short)0);
  }
  __syncthreads();

  float c_reg[4] = {0.f, 0.f, 0.f, 0.f};
  const v8f vzero = {0.f, 0.f, 0.f, 0.f, 0.f, 0.f, 0.f, 0.f};

  for (int t = 0; t < Ssz; ++t) {
    const __bf16* hrd = hbf + (size_t)(t & 1) * (Bsz * Hsz);
    __bf16*       hwr = hbf + (size_t)((t + 1) & 1) * (Bsz * Hsz);

    v8f acc0 = vzero, acc1 = vzero;
    for (int kt = 0; kt < Hsz / 32; ++kt) {
      const int k0 = kt * 32;
      const v16bf a  = ld_frag_k(hrd + (size_t)(wm * 16) * Hsz + k0, Hsz);
      const v16bf b0 = ld_frag_k(Wsh + (size_t)(q * 32) * WLD + k0, WLD);
      const v16bf b1 = ld_frag_k(Wsh + (size_t)(q * 32 + 16) * WLD + k0, WLD);
      acc0 = wmma_bf16(a, b0, acc0);
      acc1 = wmma_bf16(a, b1, acc1);
    }
#pragma unroll
    for (int v = 0; v < 8; ++v) {            // gates -> LDS [m][gate*32+local]
      const int m = wm * 16 + (lane >> 4) * 8 + v;
      const int jl = lane & 15;
      Gs[m * 132 + q * 32 + jl]      = acc0[v];
      Gs[m * 132 + q * 32 + 16 + jl] = acc1[v];
    }
    __syncthreads();
#pragma unroll
    for (int e = 0; e < 4; ++e) {            // LSTM cell + spike; c in regs
      const int eps = e * 256 + tid;
      const int m = eps >> 5;                // batch
      const int j = eps & 31;                // local hidden
      const size_t xrow = ((size_t)m * Ssz + t) * Gsz;
      const float ig = Gs[m * 132 +  0 + j] + (float)xg[xrow + 0 * Hsz + hid0 + j];
      const float fg = Gs[m * 132 + 32 + j] + (float)xg[xrow + 1 * Hsz + hid0 + j];
      const float gg = Gs[m * 132 + 64 + j] + (float)xg[xrow + 2 * Hsz + hid0 + j];
      const float og = Gs[m * 132 + 96 + j] + (float)xg[xrow + 3 * Hsz + hid0 + j];
      const float c = sigmoidf_(fg) * c_reg[e] + sigmoidf_(ig) * tanhf(gg);
      c_reg[e] = c;
      const float h = sigmoidf_(og) * tanhf(c);
      hwr[(size_t)m * Hsz + hid0 + j] = (__bf16)h;
      spk[((size_t)m * Ssz + t) * Hsz + hid0 + j] =
          (__bf16)((h - thr) > 0.0f ? 1.0f : 0.0f);
      if (t + 1 < Ssz) __builtin_prefetch(xg + xrow + Gsz + hid0 + j, 0, 0);
    }
    // grid barrier: make h(t+1) globally visible before anyone reads it
    __threadfence();
    __syncthreads();
    if (tid == 0) {
      atomicAdd(bar, 1u);
      const unsigned target = (unsigned)(t + 1) * (unsigned)nwg;
      while (__atomic_load_n(bar, __ATOMIC_RELAXED) < target)
        __builtin_amdgcn_s_sleep(2);
    }
    __syncthreads();
    __threadfence();
  }
}

// ---- fused mixer + Dyt:  out = g*tanh(a*(spk@Wc^T + x@W2^T + bc)) + b -------
__global__ void __launch_bounds__(256)
K_mix(const __bf16* __restrict__ spk, const float* __restrict__ x,
      const __bf16* __restrict__ Wc, const __bf16* __restrict__ W2,
      const float* __restrict__ bc, const float* __restrict__ alpha_p,
      const float* __restrict__ gamma, const float* __restrict__ beta,
      float* __restrict__ out) {
  __shared__ __bf16 As[128 * 40];
  __shared__ __bf16 Bs[128 * 40];
  __shared__ float bcs[128], gms[128], bts[128];

  const int tid = threadIdx.x;
  const int m0 = blockIdx.x * 128;
  const int n0 = blockIdx.y * 128;
  if (tid < 128) { bcs[tid] = bc[n0 + tid]; gms[tid] = gamma[n0 + tid]; bts[tid] = beta[n0 + tid]; }
  const float alpha = alpha_p[0];

  const int wave = tid >> 5;
  const int wm = wave & 3;
  const int wn = wave >> 2;
  const int lane = tid & 31;

  const v8f vzero = {0.f, 0.f, 0.f, 0.f, 0.f, 0.f, 0.f, 0.f};
  v8f acc[2][4];
#pragma unroll
  for (int i = 0; i < 2; ++i)
#pragma unroll
    for (int j = 0; j < 4; ++j) acc[i][j] = vzero;

  for (int kt = 0; kt < 64; ++kt) {          // K = H (spk/Wc) then D (x/W2)
    if (kt < 32) {
      const int k0 = kt * 32;
#pragma unroll
      for (int i = 0; i < 2; ++i) {
        const int s = tid + i * 256;
        const int row = s >> 2, kk = (s & 3) * 8;
        *(uint4*)(As + row * 40 + kk) =
            *(const uint4*)(spk + (size_t)(m0 + row) * Hsz + k0 + kk);
      }
#pragma unroll
      for (int i = 0; i < 2; ++i) {
        const int s = tid + i * 256;
        const int row = s >> 2, kk = (s & 3) * 8;
        *(uint4*)(Bs + row * 40 + kk) =
            *(const uint4*)(Wc + (size_t)(n0 + row) * Hsz + k0 + kk);
      }
    } else {
      const int k0 = (kt - 32) * 32;
#pragma unroll
      for (int i = 0; i < 4; ++i) {
        const int s = tid + i * 256;
        const int row = s >> 3, kk = (s & 7) * 4;
        const float4 v = *(const float4*)(x + (size_t)(m0 + row) * Dsz + k0 + kk);
        union { __bf16 h[4]; uint2 u; } pk;
        pk.h[0] = (__bf16)v.x; pk.h[1] = (__bf16)v.y;
        pk.h[2] = (__bf16)v.z; pk.h[3] = (__bf16)v.w;
        *(uint2*)(As + row * 40 + kk) = pk.u;
      }
#pragma unroll
      for (int i = 0; i < 2; ++i) {
        const int s = tid + i * 256;
        const int row = s >> 2, kk = (s & 3) * 8;
        *(uint4*)(Bs + row * 40 + kk) =
            *(const uint4*)(W2 + (size_t)(n0 + row) * Dsz + k0 + kk);
      }
    }
    __syncthreads();
    v16bf a[2], b[4];
#pragma unroll
    for (int i = 0; i < 2; ++i) a[i] = ld_frag_k(As + (wm * 32 + i * 16) * 40, 40);
#pragma unroll
    for (int j = 0; j < 4; ++j) b[j] = ld_frag_k(Bs + (wn * 64 + j * 16) * 40, 40);
#pragma unroll
    for (int i = 0; i < 2; ++i)
#pragma unroll
      for (int j = 0; j < 4; ++j) acc[i][j] = wmma_bf16(a[i], b[j], acc[i][j]);
    __syncthreads();
  }
#pragma unroll
  for (int i = 0; i < 2; ++i)
#pragma unroll
    for (int j = 0; j < 4; ++j) {
      const int mb = m0 + wm * 32 + i * 16 + (lane >> 4) * 8;
      const int nl = wn * 64 + j * 16 + (lane & 15);
      const float bb = bcs[nl], gm = gms[nl], bt = bts[nl];
#pragma unroll
      for (int v = 0; v < 8; ++v) {
        const float pre = acc[i][j][v] + bb;
        out[(size_t)(mb + v) * Hsz + (n0 + nl)] = gm * tanhf(alpha * pre) + bt;
      }
    }
}

// ---------------------------------------------------------------------------
extern "C" void kernel_launch(void* const* d_in, const int* in_sizes, int n_in,
                              void* d_out, int out_size, void* d_ws, size_t ws_size,
                              hipStream_t stream) {
  (void)in_sizes; (void)n_in; (void)out_size;
  if (ws_size < WS_NEED) return;

  const float* x     = (const float*)d_in[0];
  const float* W_ih  = (const float*)d_in[1];
  const float* W_hh  = (const float*)d_in[2];
  const float* b_ih  = (const float*)d_in[3];
  const float* b_hh  = (const float*)d_in[4];
  const float* thr   = (const float*)d_in[5];
  /* d_in[6] = sg_alpha: backward-only, unused in forward */
  const float* lin_w = (const float*)d_in[7];
  const float* lin_b = (const float*)d_in[8];
  const float* mix_w = (const float*)d_in[9];
  const float* mix_b = (const float*)d_in[10];
  const float* dyt_a = (const float*)d_in[11];
  const float* dyt_g = (const float*)d_in[12];
  const float* dyt_b = (const float*)d_in[13];
  float* out = (float*)d_out;

  char* ws = (char*)d_ws;
  __bf16*   xg_bf  = (__bf16*)(ws + OFF_XG);
  __bf16*   spk_bf = (__bf16*)(ws + OFF_SPK);
  __bf16*   Wih_bf = (__bf16*)(ws + OFF_WIH);
  __bf16*   Whh_p  = (__bf16*)(ws + OFF_WHH);
  __bf16*   Wc_bf  = (__bf16*)(ws + OFF_WC);
  __bf16*   W2_bf  = (__bf16*)(ws + OFF_W2);
  float*    bc     = (float*)(ws + OFF_BC);
  __bf16*   h_bf   = (__bf16*)(ws + OFF_H);
  unsigned* bar    = (unsigned*)(ws + OFF_BAR);

  K_init<<<(2 * Bsz * Hsz + 255) / 256, 256, 0, stream>>>(h_bf, bar);
  K_cvt<<<2048, 256, 0, stream>>>(W_ih, Wih_bf, Gsz * Dsz);
  K_prep_whh<<<(NWG * ROWS_WG * 1024 + 255) / 256, 256, 0, stream>>>(W_hh, Whh_p);
  K_wc<<<dim3(Hsz / 16, Hsz / 16), 256, 0, stream>>>(mix_w, lin_w, Wc_bf);
  K_bc<<<(Hsz + 255) / 256, 256, 0, stream>>>(mix_w, lin_b, mix_b, bc);
  K_cvt_mixw2<<<(Hsz * Dsz + 255) / 256, 256, 0, stream>>>(mix_w, W2_bf);

  K_xg<<<dim3(Mtot / 128, Gsz / 128), 256, 0, stream>>>(x, Wih_bf, b_ih, b_hh, xg_bf);
  K_scan<<<NWG, 256, SCAN_SMEM, stream>>>(xg_bf, Whh_p, h_bf, spk_bf, thr, bar, NWG);
  K_mix<<<dim3(Mtot / 128, Hsz / 128), 256, 0, stream>>>(spk_bf, x, Wc_bf, W2_bf,
                                                         bc, dyt_a, dyt_g, dyt_b, out);
}
